// LSTMClassifier_35673998361271
// MI455X (gfx1250) — compile-verified, matrix operand order checked by reading, not run on hardware
//
#include <hip/hip_runtime.h>

// ---------------------------------------------------------------------------
// LSTM classifier for MI455X (gfx1250, wave32, WMMA).
// Recurrence runs as a single persistent workgroup (16 waves / 1 WGP) using
// v_wmma_f32_16x16x64_fp8_fp8 with W_hh (fp8, 256KB) resident in LDS.
// Each wave owns 2 M-tiles x 8 gate N-tiles: every B fragment feeds two
// WMMAs (halves LDS B traffic), and fragment loads are software-pipelined
// (double-buffered B, next-k A loaded under current-k WMMAs).
// ---------------------------------------------------------------------------

#define B64 64
#define TIN 2048
#define CIN 128
#define HDIM 256
#define NOUT 10
#define TP 2046  // conv 'valid' output length

typedef int   v8i __attribute__((ext_vector_type(8)));
typedef float v8f __attribute__((ext_vector_type(8)));

#define A_SCALE 4.0f          // activation fp8 scale
#define B_SCALE 16.0f         // weight fp8 scale
#define GATE_INV (1.0f / (4.0f * 16.0f))

// f32 -> fp8 e4m3 (round-to-nearest, clamp to 448, flush sub-2^-6 to 0)
__device__ __forceinline__ unsigned to_fp8(float x) {
  unsigned u = __float_as_uint(x);
  unsigned s = (u >> 24) & 0x80u;
  unsigned a = u & 0x7FFFFFFFu;
  if (a > 0x43E00000u) a = 0x43E00000u;   // 448.0 = e4m3 max
  if ((a >> 23) < 121u) return s;         // below 2^-6 -> 0
  a += 0x00080000u;                       // round into 3-bit mantissa
  unsigned e = a >> 23;
  unsigned m = (a >> 20) & 7u;
  return s | ((e - 120u) << 3) | m;
}

// K index of byte b of dword d in an 8-bit WMMA fragment (ISA 7.12.2 layout):
// V0: K=0-3 / 8-11 (hi lanes), V1: 4-7 / 12-15, V2: 16-19/24-27, ... V4-7: +32
__device__ __forceinline__ int k_of(int d, int b, int hi) {
  return ((d & 1) << 2) | (((d >> 1) & 1) << 4) | ((d >> 2) << 5) | (hi << 3) | b;
}

__device__ __forceinline__ float sigm(float x) { return 1.0f / (1.0f + __expf(-x)); }
__device__ __forceinline__ float tanh_fast(float x) {
  float e = __expf(-2.0f * x);
  return (1.0f - e) / (1.0f + e);
}

// --------------------------- K0: per-(b,ci) L2 norm over T ------------------
__global__ void norm_kernel(const float* __restrict__ x, float* __restrict__ invn) {
  int b = blockIdx.x, ci = threadIdx.x;
  float s = 0.0f;
  for (int t = 0; t < TIN; ++t) {
    float v = x[(b * TIN + t) * CIN + ci];
    s = fmaf(v, v, s);
  }
  invn[b * CIN + ci] = 1.0f / fmaxf(sqrtf(s), 1e-12f);
}

// --------------------------- K1: conv1d (valid, K=3) ------------------------
// y layout: [t'][b][co]
__global__ void conv_kernel(const float* __restrict__ x, const float* __restrict__ invn,
                            const float* __restrict__ Wc, const float* __restrict__ bc,
                            float* __restrict__ y) {
  extern __shared__ float sm[];
  float* sW = sm;              // 64 rows * 387 (padded to kill bank conflicts)
  float* sX = sm + 64 * 387;   // 34 * 128 normalized input slice
  const int t0 = blockIdx.x * 32;
  const int b = blockIdx.y;
  const int tid = threadIdx.x;
  for (int i = tid; i < 64 * 128 * 3; i += 256) sW[(i / 384) * 387 + (i % 384)] = Wc[i];
  for (int i = tid; i < 34 * 128; i += 256) {
    int tt = i >> 7, ci = i & 127, tg = t0 + tt;
    sX[i] = (tg < TIN) ? x[(b * TIN + tg) * CIN + ci] * invn[b * CIN + ci] : 0.0f;
  }
  __syncthreads();
  const int co = tid & 63, tl = tid >> 6;
  const float* wr = &sW[co * 387];
  float acc[8];
  float bcv = bc[co];
#pragma unroll
  for (int s = 0; s < 8; ++s) acc[s] = bcv;
  const int tloc = tl * 8;
  for (int ci = 0; ci < 128; ++ci) {
    float w0 = wr[ci * 3], w1 = wr[ci * 3 + 1], w2 = wr[ci * 3 + 2];
    float xv[10];
#pragma unroll
    for (int u = 0; u < 10; ++u) xv[u] = sX[(tloc + u) * 128 + ci];
#pragma unroll
    for (int s = 0; s < 8; ++s)
      acc[s] = fmaf(xv[s], w0, fmaf(xv[s + 1], w1, fmaf(xv[s + 2], w2, acc[s])));
  }
#pragma unroll
  for (int s = 0; s < 8; ++s) {
    int tp = t0 + tloc + s;
    if (tp < TP) y[(tp * 64 + b) * 64 + co] = acc[s];
  }
}

// --------------------------- K2: BN batch statistics ------------------------
__global__ void bnstats_kernel(const float* __restrict__ y, float* __restrict__ stats) {
  __shared__ float rs[256], rq[256];
  int co = blockIdx.x, tid = threadIdx.x;
  float s = 0.0f, q = 0.0f;
  for (int i = tid; i < B64 * TP; i += 256) {   // i = t*64 + b
    float v = y[i * 64 + co];
    s += v;
    q = fmaf(v, v, q);
  }
  rs[tid] = s; rq[tid] = q;
  __syncthreads();
  for (int off = 128; off > 0; off >>= 1) {
    if (tid < off) { rs[tid] += rs[tid + off]; rq[tid] += rq[tid + off]; }
    __syncthreads();
  }
  if (tid == 0) { stats[co] = rs[0]; stats[64 + co] = rq[0]; }
}

// ---------- K3: BN apply + ReLU + fp8 quantize + swizzle to A-fragments -----
// Fragment dword order (lane-contiguous): idx = m*256 + lane*8 + d
__global__ void pack_x_kernel(const float* __restrict__ y, const float* __restrict__ stats,
                              const float* __restrict__ gamma, const float* __restrict__ beta,
                              unsigned* __restrict__ gx8) {
  const int t = blockIdx.x;
  const int tid = threadIdx.x;
  const float invN = 1.0f / (float)(B64 * TP);
#pragma unroll
  for (int rep = 0; rep < 4; ++rep) {
    int idx = tid + 256 * rep;
    int m = idx >> 8, lane = (idx >> 3) & 31, d = idx & 7;
    int r = m * 16 + (lane & 15);   // batch row (GEMM M)
    int hi = lane >> 4;
    unsigned pk = 0u;
#pragma unroll
    for (int b2 = 0; b2 < 4; ++b2) {
      int co = k_of(d, b2, hi);     // conv channel (GEMM K)
      float mean = stats[co] * invN;
      float var = fmaf(-mean, mean, stats[64 + co] * invN);
      float sc = gamma[co] * rsqrtf(var + 1e-5f);
      float sh = fmaf(-mean, sc, beta[co]);
      float v = fmaxf(fmaf(y[(t * 64 + r) * 64 + co], sc, sh), 0.0f);
      pk |= to_fp8(v * A_SCALE) << (8 * b2);
    }
    gx8[t * 1024 + idx] = pk;
  }
}

// ---------- K4: quantize + swizzle W_hh^T / W_ih^T into B-fragments ---------
// whh8: [kt][nTile][lane*8+d]  wih8: [nTile][lane*8+d]
__global__ void prep_weights(const float* __restrict__ W_hh, const float* __restrict__ W_ih,
                             unsigned* __restrict__ whh8, unsigned* __restrict__ wih8) {
  int idx = blockIdx.x * 256 + threadIdx.x;
  if (idx < 65536) {
    int lane = (idx >> 3) & 31, d = idx & 7, n = (idx >> 8) & 63, kt = idx >> 14;
    int hi = lane >> 4, ncol = n * 16 + (lane & 15);
    unsigned pk = 0u;
#pragma unroll
    for (int b2 = 0; b2 < 4; ++b2) {
      int k = kt * 64 + k_of(d, b2, hi);
      pk |= to_fp8(W_hh[ncol * HDIM + k] * B_SCALE) << (8 * b2);
    }
    whh8[idx] = pk;
  } else {
    int i2 = idx - 65536;
    if (i2 < 16384) {
      int lane = (i2 >> 3) & 31, d = i2 & 7, n = i2 >> 8;
      int hi = lane >> 4, ncol = n * 16 + (lane & 15);
      unsigned pk = 0u;
#pragma unroll
      for (int b2 = 0; b2 < 4; ++b2) {
        int k = k_of(d, b2, hi);
        pk |= to_fp8(W_ih[ncol * 64 + k] * B_SCALE) << (8 * b2);
      }
      wih8[i2] = pk;
    }
  }
}

// ---------------- K5: persistent FP8-WMMA LSTM (one workgroup) --------------
// 512 threads = 16 waves (4/SIMD -> double VGPR budget, no spills).
// Wave w: mp = w&1 -> M-tiles {2mp, 2mp+1}; g = w>>1 -> h cols [32g, 32g+32),
// gate N-tiles {q*16 + 2g + j}. Every B fragment feeds 2 WMMAs.
// K passes: 1x x-pass (W_ih from global, WGP$-resident) + 4x h-pass (LDS),
// with double-buffered B fragments and pipelined next-k A fragments.
__global__ __launch_bounds__(512) void lstm_kernel(
    const unsigned* __restrict__ whh8, const unsigned* __restrict__ wih8,
    const unsigned* __restrict__ gx8, const float* __restrict__ b_ih,
    const float* __restrict__ b_hh, float* __restrict__ hfin) {
  extern __shared__ unsigned lds[];
  unsigned* lds_whh = lds;            // 65536 dwords = 256 KB
  unsigned* lds_h   = lds + 65536;    // 2 * 4096 dwords = 32 KB
  const int tid = threadIdx.x;
  const int lane = tid & 31;
  const int wave = tid >> 5;
  const int mp = wave & 1;   // M-pair: tiles 2mp, 2mp+1
  const int g = wave >> 1;   // h-column group

  for (int i = tid; i < 65536; i += 512) lds_whh[i] = whh8[i];
  for (int i = tid; i < 8192; i += 512) lds_h[i] = 0u;   // h_0 = 0 (both buffers)

  float bias[4][2];
#pragma unroll
  for (int q = 0; q < 4; ++q)
#pragma unroll
    for (int j = 0; j < 2; ++j) {
      int col = q * 256 + g * 32 + j * 16 + (lane & 15);
      bias[q][j] = b_ih[col] + b_hh[col];
    }

  v8f cst[2][2] = {};   // f32 cell state [p][j], register-resident
  __syncthreads();

  const unsigned* gxp = gx8 + mp * 512 + lane * 8;         // x frags (p=0 at +0, p=1 at +256)
  const unsigned* wip = wih8 + g * 2 * 256 + lane * 8;     // W_ih frag base
  const unsigned* whp = lds_whh + g * 2 * 256 + lane * 8;  // W_hh frag base

  int cur = 0;
  for (int t = 0; t < TP; ++t) {
    // Forbid hoisting loop-invariant global loads (they would spill): W_ih is
    // re-read each step and hits the WGP cache.
    asm volatile("" ::: "memory");

    const unsigned* hbase = lds_h + cur * 4096 + mp * 2048 + lane * 8;

    v8f acc[4][2][2];  // [q][j][p]
    // ---- x pass (K 0..63): A = x_t fragments, B = W_ih^T (global/WGP$) ----
    {
      v8i ax0 = *(const v8i*)(gxp + t * 1024);
      v8i ax1 = *(const v8i*)(gxp + t * 1024 + 256);
      if (t + 1 < TP) {
        __builtin_prefetch(gxp + (t + 1) * 1024, 0, 1);
        __builtin_prefetch(gxp + (t + 1) * 1024 + 256, 0, 1);
      }
      const v8f zero = {};
      v8i bcur = *(const v8i*)(wip);
#pragma unroll
      for (int u = 0; u < 8; ++u) {
        v8i bnxt = bcur;
        if (u < 7) {
          int u1 = u + 1;
          bnxt = *(const v8i*)(wip + ((u1 >> 1) * 16 + (u1 & 1)) * 256);
        }
        const int q = u >> 1, j = u & 1;
        acc[q][j][0] = __builtin_amdgcn_wmma_f32_16x16x64_fp8_fp8(
            ax0, bcur, (short)0, zero, false, false);
        acc[q][j][1] = __builtin_amdgcn_wmma_f32_16x16x64_fp8_fp8(
            ax1, bcur, (short)0, zero, false, false);
        bcur = bnxt;
      }
    }
    // ---- h passes (K 64..319): A = h_t fragments (LDS), B = W_hh^T (LDS) ----
    {
      v8i ah0 = *(const v8i*)(hbase);           // (m=2mp,   kt=0)
      v8i ah1 = *(const v8i*)(hbase + 1024);    // (m=2mp+1, kt=0)
#pragma unroll
      for (int kt = 0; kt < 4; ++kt) {
        v8i ah0n = ah0, ah1n = ah1;
        if (kt < 3) {  // next k-step's A fragments load under current WMMAs
          ah0n = *(const v8i*)(hbase + (kt + 1) * 256);
          ah1n = *(const v8i*)(hbase + 1024 + (kt + 1) * 256);
        }
        v8i bcur = *(const v8i*)(whp + (kt * 64) * 256);
#pragma unroll
        for (int u = 0; u < 8; ++u) {
          v8i bnxt = bcur;
          if (u < 7) {
            int u1 = u + 1;
            bnxt = *(const v8i*)(whp + (kt * 64 + (u1 >> 1) * 16 + (u1 & 1)) * 256);
          }
          const int q = u >> 1, j = u & 1;
          acc[q][j][0] = __builtin_amdgcn_wmma_f32_16x16x64_fp8_fp8(
              ah0, bcur, (short)0, acc[q][j][0], false, false);
          acc[q][j][1] = __builtin_amdgcn_wmma_f32_16x16x64_fp8_fp8(
              ah1, bcur, (short)0, acc[q][j][1], false, false);
          bcur = bnxt;
        }
        ah0 = ah0n;
        ah1 = ah1n;
      }
    }

    // gates -> c,h update; write h_{t+1} fp8 fragments into the other buffer
    const int nb = cur ^ 1;
    unsigned char* hdst = (unsigned char*)(lds_h + nb * 4096);
#pragma unroll
    for (int p = 0; p < 2; ++p) {
      const int m = 2 * mp + p;
      const int rbase = m * 16 + ((lane >> 4) << 3);  // C/D layout: M = v (+8 hi lanes)
#pragma unroll
      for (int j = 0; j < 2; ++j) {
        const int k = g * 32 + j * 16 + (lane & 15);  // h column (next-step K index)
        const int kt2 = k >> 6, kk = k & 63;
        const int d2 = ((kk >> 2) & 1) | ((kk >> 3) & 2) | ((kk >> 3) & 4);
        const int hi2 = (kk >> 3) & 1;
        // byte addr = (((m*4+kt2)*256 + lane2*8 + d2) << 2) + (kk&3)
        const unsigned base = (((unsigned)(m * 4 + kt2) * 256u + (unsigned)d2) << 2)
                              + (unsigned)(kk & 3);
#pragma unroll
        for (int v = 0; v < 8; ++v) {
          float gi = fmaf(acc[0][j][p][v], GATE_INV, bias[0][j]);
          float gf = fmaf(acc[1][j][p][v], GATE_INV, bias[1][j]);
          float gg = fmaf(acc[2][j][p][v], GATE_INV, bias[2][j]);
          float go = fmaf(acc[3][j][p][v], GATE_INV, bias[3][j]);
          float iv = sigm(gi), fv = sigm(gf), cv = tanh_fast(gg), ov = sigm(go);
          float cc = fmaf(fv, cst[p][j][v], iv * cv);
          cst[p][j][v] = cc;
          float hv = ov * tanh_fast(cc);
          int r = rbase + v;
          int lane2 = (r & 15) + (hi2 << 4);
          hdst[base + (unsigned)lane2 * 32u] = (unsigned char)to_fp8(hv * A_SCALE);
          if (t == TP - 1) hfin[r * HDIM + k] = hv;
        }
      }
    }
    __syncthreads();
    cur = nb;
  }
}

// --------------------------- K6: final linear head --------------------------
__global__ void head_kernel(const float* __restrict__ hfin, const float* __restrict__ W_lin,
                            const float* __restrict__ b_lin, float* __restrict__ out) {
  int tid = threadIdx.x;
  if (tid >= B64 * NOUT) return;
  int b = tid / NOUT, o = tid % NOUT;
  float acc = b_lin[o];
  for (int j = 0; j < HDIM; ++j) acc = fmaf(hfin[b * HDIM + j], W_lin[o * HDIM + j], acc);
  out[tid] = acc;
}

// ---------------------------------------------------------------------------
extern "C" void kernel_launch(void* const* d_in, const int* in_sizes, int n_in,
                              void* d_out, int out_size, void* d_ws, size_t ws_size,
                              hipStream_t stream) {
  const float* input  = (const float*)d_in[0];
  // d_in[1] = r (unused by the reference)
  const float* W_conv = (const float*)d_in[2];
  const float* b_conv = (const float*)d_in[3];
  const float* gamma  = (const float*)d_in[4];
  const float* beta   = (const float*)d_in[5];
  const float* W_ih   = (const float*)d_in[6];
  const float* W_hh   = (const float*)d_in[7];
  const float* b_ih   = (const float*)d_in[8];
  const float* b_hh   = (const float*)d_in[9];
  const float* W_lin  = (const float*)d_in[10];
  const float* b_lin  = (const float*)d_in[11];
  float* out = (float*)d_out;

  char* ws = (char*)d_ws;
  unsigned* whh8 = (unsigned*)(ws + 0);          // 256 KB
  unsigned* wih8 = (unsigned*)(ws + 262144);     // 64 KB
  unsigned* gx8  = (unsigned*)(ws + 327680);     // 8.0 MB  (TP*4KB fp8 x-fragments)
  float*    y    = (float*)(ws + 8708096);       // 32 MB   conv output [t'][b][co]
  float*    invn = (float*)(ws + 42229760);      // 32 KB
  float*    stats = (float*)(ws + 42262528);     // sums / sumsq per channel
  float*    hfin = (float*)(ws + 42262784);      // 64 KB   final hidden state f32

  norm_kernel<<<64, 128, 0, stream>>>(input, invn);
  prep_weights<<<320, 256, 0, stream>>>(W_hh, W_ih, whh8, wih8);
  conv_kernel<<<dim3(64, 64), 256, 116480, stream>>>(input, invn, W_conv, b_conv, y);
  bnstats_kernel<<<64, 256, 0, stream>>>(y, stats);
  pack_x_kernel<<<TP, 256, 0, stream>>>(y, stats, gamma, beta, gx8);
  lstm_kernel<<<1, 512, 294912, stream>>>(whh8, wih8, gx8, b_ih, b_hh, hfin);
  head_kernel<<<1, 640, 0, stream>>>(hfin, W_lin, b_lin, out);

  (void)in_sizes; (void)n_in; (void)out_size; (void)ws_size;
}
